// MambaVisionMixer_19387482374998
// MI455X (gfx1250) — compile-verified, matrix operand order checked by reading, not run on hardware
//
#include <hip/hip_runtime.h>
#include <hip/hip_bf16.h>

// ---------------- problem constants ----------------
#define D_MODEL  1024
#define D_STATE  16
#define D_INNER  2048
#define D_HALF   1024
#define DT_RANK  64
#define BATCH    16
#define SEQ      1024
#define M_TOTAL  (BATCH * SEQ)   // 16384 token rows

// ---------------- vector types ----------------
typedef __attribute__((ext_vector_type(16))) __bf16   v16bf;
typedef __attribute__((ext_vector_type(8)))  float    v8f;
typedef __attribute__((ext_vector_type(4)))  float    f32x4;
typedef __attribute__((ext_vector_type(4)))  unsigned u32x4;
typedef __attribute__((ext_vector_type(2)))  unsigned u32x2;

union FragBF { v16bf v; u32x4 q[2]; };
union Pack4  { __bf16 h[4]; u32x2 u; };

// ---------------- GEMM tiling ----------------
#define BM 128
#define BN 128
#define BK 32

// C[M,N] = concat(A1|A2)[M,K] * W[N,K]^T   (fp32 in/out, bf16 WMMA, f32 accum)
// Columns k < split come from A1 (row stride lda1), k >= split from A2 (lda2).
// split is always a multiple of BK, so each K-tile is wholly on one side
// (uniform scalar pointer select). LDS double-buffered; global loads for
// tile k+1 are issued before the WMMAs of tile k (latency overlap).
__global__ __launch_bounds__(256)
void mamba_gemm_bf16_wmma(const float* __restrict__ A1, const float* __restrict__ A2,
                          const float* __restrict__ W,  float* __restrict__ C,
                          int M, int N, int K, int lda1, int lda2, int split)
{
    __shared__ __bf16 ldsA[2][BM * BK];   // 2 x 8 KB
    __shared__ __bf16 ldsB[2][BN * BK];   // 2 x 8 KB

    const int tid  = threadIdx.x;
    const int lane = tid & 31;
    const int wave = tid >> 5;        // 0..7
    const int wm   = wave & 1;        // -> 64 rows of M
    const int wn   = wave >> 1;       // -> 32 cols of N
    const int l16  = lane & 15;
    const int hi   = lane >> 4;       // lane half (CDNA5 16-bit frag layout)
    const int m0   = blockIdx.y * BM;
    const int n0   = blockIdx.x * BN;

    const v8f vzero = {0.f, 0.f, 0.f, 0.f, 0.f, 0.f, 0.f, 0.f};
    v8f acc[4][2];
#pragma unroll
    for (int i = 0; i < 4; ++i)
#pragma unroll
        for (int j = 0; j < 2; ++j) acc[i][j] = vzero;

    f32x4 ra[4], rw[4];               // register staging for next tile
    const f32x4 fzero = {0.f, 0.f, 0.f, 0.f};

    // issue global loads for the K-tile starting at k0 into registers
    auto load_tiles = [&](int k0) {
        const float* Ab;
        int lda, kof;
        if (k0 < split) { Ab = A1; lda = lda1; kof = k0; }          // uniform
        else            { Ab = A2; lda = lda2; kof = k0 - split; }
#pragma unroll
        for (int i = 0; i < 4; ++i) {
            int idx = tid + 256 * i;
            int row = idx >> 3;
            int c4  = (idx & 7) << 2;
            const float* src = Ab + (long long)(m0 + row) * lda + kof + c4;
            ra[i] = *(const f32x4*)src;
            if (k0 + BK < K) __builtin_prefetch(src + BK, 0, 1);
        }
#pragma unroll
        for (int i = 0; i < 4; ++i) {
            int idx = tid + 256 * i;
            int row = idx >> 3;
            int c4  = (idx & 7) << 2;
            int n   = n0 + row;
            if (n < N) {
                const float* src = W + (long long)n * K + k0 + c4;
                rw[i] = *(const f32x4*)src;
                if (k0 + BK < K) __builtin_prefetch(src + BK, 0, 1);
            } else {
                rw[i] = fzero;
            }
        }
    };

    // convert staged registers to bf16 and write into LDS buffer `buf`
    auto store_tiles = [&](int buf) {
#pragma unroll
        for (int i = 0; i < 4; ++i) {
            int idx = tid + 256 * i;
            int row = idx >> 3;
            int c4  = (idx & 7) << 2;
            Pack4 p;
            p.h[0] = (__bf16)ra[i].x; p.h[1] = (__bf16)ra[i].y;
            p.h[2] = (__bf16)ra[i].z; p.h[3] = (__bf16)ra[i].w;
            *(u32x2*)&ldsA[buf][row * BK + c4] = p.u;
        }
#pragma unroll
        for (int i = 0; i < 4; ++i) {
            int idx = tid + 256 * i;
            int row = idx >> 3;
            int c4  = (idx & 7) << 2;
            Pack4 p;
            p.h[0] = (__bf16)rw[i].x; p.h[1] = (__bf16)rw[i].y;
            p.h[2] = (__bf16)rw[i].z; p.h[3] = (__bf16)rw[i].w;
            *(u32x2*)&ldsB[buf][row * BK + c4] = p.u;
        }
    };

    const int nk = K / BK;
    load_tiles(0);
    store_tiles(0);
    __syncthreads();

    for (int kt = 0; kt < nk; ++kt) {
        const int cur = kt & 1;
        if (kt + 1 < nk) load_tiles((kt + 1) * BK);   // global loads in flight

        // ---- fragment gather per CDNA5 16-bit layout:
        //      elements 0..7  <- K = hi*8 + 0..7
        //      elements 8..15 <- K = 16 + hi*8 + 0..7   (2x ds_load_b128)
        FragBF a[4], b[2];
#pragma unroll
        for (int mf = 0; mf < 4; ++mf) {
            int r = wm * 64 + mf * 16 + l16;
            a[mf].q[0] = *(const u32x4*)&ldsA[cur][r * BK + hi * 8];
            a[mf].q[1] = *(const u32x4*)&ldsA[cur][r * BK + 16 + hi * 8];
        }
#pragma unroll
        for (int nf = 0; nf < 2; ++nf) {
            int r = wn * 32 + nf * 16 + l16;
            b[nf].q[0] = *(const u32x4*)&ldsB[cur][r * BK + hi * 8];
            b[nf].q[1] = *(const u32x4*)&ldsB[cur][r * BK + 16 + hi * 8];
        }
#pragma unroll
        for (int mf = 0; mf < 4; ++mf)
#pragma unroll
            for (int nf = 0; nf < 2; ++nf)
                acc[mf][nf] = __builtin_amdgcn_wmma_f32_16x16x32_bf16(
                    false, a[mf].v, false, b[nf].v, (short)0, acc[mf][nf], false, false);

        if (kt + 1 < nk) store_tiles(cur ^ 1);        // waits loads, cvt, ds_store
        __syncthreads();
    }

    // ---- epilogue: C/D layout -> M = hi*8 + r, N = l16 ----
#pragma unroll
    for (int mf = 0; mf < 4; ++mf)
#pragma unroll
        for (int nf = 0; nf < 2; ++nf) {
            int n = n0 + wn * 32 + nf * 16 + l16;
            if (n < N) {
#pragma unroll
                for (int r = 0; r < 8; ++r) {
                    int m = m0 + wm * 64 + mf * 16 + hi * 8 + r;
                    C[(long long)m * N + n] = acc[mf][nf][r];
                }
            }
        }
}

// ---- depthwise conv K=4 (pad left 1, right 2) + SiLU; xz layout (B, L, 2048) ----
__global__ __launch_bounds__(256)
void mamba_conv_silu(const float* __restrict__ xz, const float* __restrict__ w,
                     float* __restrict__ out, int off)
{
    long long idx = (long long)blockIdx.x * 256 + threadIdx.x;   // over B*L*D_HALF
    int c = (int)(idx & (D_HALF - 1));
    long long bl = idx >> 10;            // b*L + l
    int l = (int)(bl & (SEQ - 1));
    long long bb = bl >> 10;
    float acc = 0.f;
#pragma unroll
    for (int k = 0; k < 4; ++k) {
        int ll = l - 1 + k;
        if ((unsigned)ll < (unsigned)SEQ)
            acc = fmaf(xz[((bb * SEQ + ll) * D_INNER) + off + c], w[c * 4 + k], acc);
    }
    out[bl * D_HALF + c] = acc / (1.f + __expf(-acc));   // silu
}

// ---- delta = softplus(raw + bias) in place; layout (M, D_HALF) ----
__global__ __launch_bounds__(256)
void mamba_softplus_bias(float* __restrict__ delta, const float* __restrict__ bias)
{
    long long idx = (long long)blockIdx.x * 256 + threadIdx.x;
    int d = (int)(idx & (D_HALF - 1));
    float v = delta[idx] + bias[d];
    delta[idx] = (v > 20.f) ? v : log1pf(__expf(v));
}

// ---- selective scan: one lane per (b, d) channel; h[16] in registers.
//      B_t/C_t broadcast through LDS, 8 timesteps per fill (256 threads load
//      8 x 32 floats -> one barrier pair per 8 steps). y overwrites delta. ----
#define T_CHUNK 8
__global__ __launch_bounds__(256)
void mamba_scan(const float* __restrict__ xc,    // (B*L, D_HALF) silu(conv(x))
                float* __restrict__ delta,       // in: delta, out: y
                const float* __restrict__ xdbl,  // (B*L, 96): [dtr(64) | B(16) | C(16)]
                const float* __restrict__ A_log, // (D_HALF, 16)
                const float* __restrict__ Dvec)  // (D_HALF)
{
    const int tid = threadIdx.x;
    const int blocksPerB = D_HALF / 256;                  // 4
    const int b = blockIdx.x / blocksPerB;
    const int d = (blockIdx.x % blocksPerB) * 256 + tid;

    __shared__ float sBC[T_CHUNK][2 * D_STATE];

    float A[D_STATE];
#pragma unroll
    for (int n = 0; n < D_STATE; ++n) A[n] = -__expf(A_log[d * D_STATE + n]);
    const float Dd = Dvec[d];

    float h[D_STATE];
#pragma unroll
    for (int n = 0; n < D_STATE; ++n) h[n] = 0.f;

    const int ldTT = tid >> 5;          // 0..7   : timestep within chunk
    const int ldE  = tid & 31;          // 0..31  : B(16)|C(16) element

    for (int t0 = 0; t0 < SEQ; t0 += T_CHUNK) {
        // cooperative fill: 256 threads <-> 8 steps x 32 floats, one each
        sBC[ldTT][ldE] = xdbl[((long long)b * SEQ + t0 + ldTT) * 96 + DT_RANK + ldE];
        __syncthreads();

#pragma unroll
        for (int j = 0; j < T_CHUNK; ++j) {
            const long long base = (long long)b * SEQ + t0 + j;
            const float dt  = delta[base * D_HALF + d];
            const float xv  = xc[base * D_HALF + d];
            const float dtx = dt * xv;
            float y = 0.f;
#pragma unroll
            for (int n = 0; n < D_STATE; ++n) {
                float dA = __expf(dt * A[n]);
                h[n] = fmaf(h[n], dA, dtx * sBC[j][n]);
                y = fmaf(h[n], sBC[j][D_STATE + n], y);
            }
            delta[base * D_HALF + d] = fmaf(Dd, xv, y);
        }
        __syncthreads();
    }
}

// ---------------- host launcher ----------------
extern "C" void kernel_launch(void* const* d_in, const int* in_sizes, int n_in,
                              void* d_out, int out_size, void* d_ws, size_t ws_size,
                              hipStream_t stream)
{
    const float* hs   = (const float*)d_in[0];  // (B, L, 1024)
    const float* inpw = (const float*)d_in[1];  // (2048, 1024)
    const float* cxw  = (const float*)d_in[2];  // (1024, 1, 4)
    const float* czw  = (const float*)d_in[3];  // (1024, 1, 4)
    const float* xpw  = (const float*)d_in[4];  // (96, 1024)
    const float* dtw  = (const float*)d_in[5];  // (1024, 64)
    const float* dtb  = (const float*)d_in[6];  // (1024)
    const float* alog = (const float*)d_in[7];  // (1024, 16)
    const float* dvec = (const float*)d_in[8];  // (1024)
    const float* outw = (const float*)d_in[9];  // (1024, 2048)
    float* out = (float*)d_out;                 // (B, L, 1024)

    // scratch layout (floats)
    float* ws    = (float*)d_ws;
    float* xz    = ws;                                     // M * 2048
    float* xc    = xz   + (size_t)M_TOTAL * D_INNER;       // M * 1024
    float* zc    = xc   + (size_t)M_TOTAL * D_HALF;        // M * 1024
    float* xdbl  = zc   + (size_t)M_TOTAL * D_HALF;        // M * 96
    float* delta = xdbl + (size_t)M_TOTAL * 96;            // M * 1024 (becomes y)

    dim3 blk(256);
    const long long nel = (long long)M_TOTAL * D_HALF;

    // 1) in_proj: xz = hs @ in_proj_w^T       (M x 2048, K=1024)
    mamba_gemm_bf16_wmma<<<dim3(D_INNER / BN, M_TOTAL / BM), blk, 0, stream>>>(
        hs, hs, inpw, xz, M_TOTAL, D_INNER, D_MODEL, D_MODEL, D_MODEL, D_MODEL);

    // 2) depthwise conv + SiLU for x and z branches
    mamba_conv_silu<<<(unsigned)(nel / 256), blk, 0, stream>>>(xz, cxw, xc, 0);
    mamba_conv_silu<<<(unsigned)(nel / 256), blk, 0, stream>>>(xz, czw, zc, D_HALF);

    // 3) x_proj: xdbl = xc @ x_proj_w^T       (M x 96, K=1024; N guarded)
    mamba_gemm_bf16_wmma<<<dim3(1, M_TOTAL / BM), blk, 0, stream>>>(
        xc, xc, xpw, xdbl, M_TOTAL, DT_RANK + 2 * D_STATE, D_HALF, D_HALF, D_HALF, D_HALF);

    // 4) dt_proj: delta_raw = xdbl[:, :64] @ dt_proj_w^T   (M x 1024, K=64, lda=96)
    mamba_gemm_bf16_wmma<<<dim3(D_HALF / BN, M_TOTAL / BM), blk, 0, stream>>>(
        xdbl, xdbl, dtw, delta, M_TOTAL, D_HALF, DT_RANK, 96, 96, DT_RANK);

    // 5) delta = softplus(delta_raw + dt_bias)
    mamba_softplus_bias<<<(unsigned)(nel / 256), blk, 0, stream>>>(delta, dtb);

    // 6) selective scan; y overwrites delta
    mamba_scan<<<BATCH * (D_HALF / 256), blk, 0, stream>>>(xc, delta, xdbl, alog, dvec);

    // 7) out_proj: out = concat(y, zc) @ out_proj_w^T   (M x 1024, K=2048, split at 1024)
    mamba_gemm_bf16_wmma<<<dim3(D_MODEL / BN, M_TOTAL / BM), blk, 0, stream>>>(
        delta, zc, outw, out, M_TOTAL, D_MODEL, D_INNER, D_HALF, D_HALF, D_HALF);
}
